// Int8LlamaDecoderLayer_30940944401186
// MI455X (gfx1250) — 16-bit and 8-bit matrix operand order repaired and checked by reading, compiled, not run on hardware
//
#include <hip/hip_runtime.h>
#include <hip/hip_bf16.h>
#include <math.h>

typedef int      v8i  __attribute__((ext_vector_type(8)));
typedef float    v8f  __attribute__((ext_vector_type(8)));
typedef _Float16 v16h __attribute__((ext_vector_type(16)));

static constexpr int Bc  = 2;
static constexpr int Sc  = 1024;
static constexpr int Hc  = 4096;
static constexpr int NHc = 32;
static constexpr int HDc = 128;
static constexpr int Ic  = 11008;
static constexpr int MT  = Bc * Sc;

__device__ __forceinline__ signed char quant8(float v) {
    int q = (int)rintf(v);
    q = q > 127 ? 127 : q;
    q = q < -128 ? -128 : q;
    return (signed char)q;
}

__global__ __launch_bounds__(256) void rmsnorm_quant_kernel(
    const float* __restrict__ x, const float* __restrict__ w,
    signed char* __restrict__ out, int Hdim) {
    const int row = blockIdx.x;
    const float* xr = x + (size_t)row * Hdim;
    float ss = 0.f;
    for (int i = threadIdx.x; i < Hdim; i += 256) { float v = xr[i]; ss += v * v; }
    __shared__ float red[256];
    red[threadIdx.x] = ss;
    __syncthreads();
    for (int s = 128; s > 0; s >>= 1) {
        if ((int)threadIdx.x < s) red[threadIdx.x] += red[threadIdx.x + s];
        __syncthreads();
    }
    const float r = rsqrtf(red[0] / (float)Hdim + 1e-6f);
    for (int i = threadIdx.x; i < Hdim; i += 256)
        out[(size_t)row * Hdim + i] = quant8(xr[i] * r * w[i]);
}

__device__ __forceinline__ void async_copy16(void* lds_dst, const void* gsrc) {
    const unsigned lds = (unsigned)(size_t)lds_dst;
    const unsigned long long ga = (unsigned long long)(size_t)gsrc;
    asm volatile("global_load_async_to_lds_b128 %0, %1, off"
                 :: "v"(lds), "v"(ga)
                 : "memory");
}
__device__ __forceinline__ void async_wait0() {
    asm volatile("s_wait_asynccnt 0" ::: "memory");
}

__global__ __launch_bounds__(256) void gemm_i8_wmma_kernel(
    const signed char* __restrict__ X, const signed char* __restrict__ W,
    const float* __restrict__ scale_ptr, const float* __restrict__ bias,
    const float* __restrict__ residual, float* __restrict__ out,
    int M, int N, int K) {
    __shared__ __align__(16) signed char sA[2][64 * 64];
    __shared__ __align__(16) signed char sB[2][64 * 64];

    const int tid  = threadIdx.x;
    const int wave = tid >> 5;
    const int lane = tid & 31;
    const int half_ = lane >> 4;
    const int m    = lane & 15;
    const int n    = lane & 15;
    const int mt   = wave & 3;
    const int nt0  = (wave >> 2) * 2;

    const int m0 = blockIdx.y * 64;
    const int n0 = blockIdx.x * 64;

    const int ldRow = tid >> 2;
    const int ldCol = (tid & 3) * 16;

    const signed char* gA = X + (size_t)(m0 + ldRow) * K + ldCol;
    const signed char* gB = W + (size_t)(n0 + ldRow) * K + ldCol;

    v8i c0 = {0, 0, 0, 0, 0, 0, 0, 0};
    v8i c1 = {0, 0, 0, 0, 0, 0, 0, 0};

    async_copy16(&sA[0][ldRow * 64 + ldCol], gA);
    async_copy16(&sB[0][ldRow * 64 + ldCol], gB);
    async_wait0();
    __syncthreads();

    int cur = 0;
    for (int k0 = 0; k0 < K; k0 += 64, cur ^= 1) {
        if (k0 + 64 < K) {
            async_copy16(&sA[cur ^ 1][ldRow * 64 + ldCol], gA + k0 + 64);
            async_copy16(&sB[cur ^ 1][ldRow * 64 + ldCol], gB + k0 + 64);
            __builtin_prefetch(gA + k0 + 128, 0, 1);
            __builtin_prefetch(gB + k0 + 128, 0, 1);
        }

        v8i a, b0, b1;
#pragma unroll
        for (int d = 0; d < 8; ++d) {
            const int ka = ((d & 3) << 2) | (half_ << 4) | ((d >> 2) << 5);
            a[d] = *(const int*)(&sA[cur][(mt * 16 + m) * 64 + ka]);
            const int kb = ((d & 3) << 2) | (half_ << 4) | ((d >> 2) << 5);
            b0[d] = *(const int*)(&sB[cur][(nt0 * 16 + n) * 64 + kb]);
            b1[d] = *(const int*)(&sB[cur][((nt0 + 1) * 16 + n) * 64 + kb]);
        }
        c0 = __builtin_amdgcn_wmma_i32_16x16x64_iu8(true, a, true, b0, c0, false, false);
        c1 = __builtin_amdgcn_wmma_i32_16x16x64_iu8(true, a, true, b1, c1, false, false);

        async_wait0();
        __syncthreads();
    }

    const float s = *scale_ptr;
#pragma unroll
    for (int d = 0; d < 8; ++d) {
        const int rowg = m0 + mt * 16 + d + (half_ << 3);
        const int col0 = n0 + nt0 * 16 + n;
        const int col1 = col0 + 16;
        const size_t i0 = (size_t)rowg * N + col0;
        const size_t i1 = (size_t)rowg * N + col1;
        float v0 = (float)c0[d] * s + bias[col0];
        float v1 = (float)c1[d] * s + bias[col1];
        if (residual) { v0 += residual[i0]; v1 += residual[i1]; }
        out[i0] = v0;
        out[i1] = v1;
    }
}

__global__ void rope_f16_kernel(const float* __restrict__ qf,
                                const float* __restrict__ kf,
                                _Float16* __restrict__ qh,
                                _Float16* __restrict__ kh) {
    const int idx = blockIdx.x;
    const int s = idx % Sc;
    const int h = (idx / Sc) % NHc;
    const int b = idx / (Sc * NHc);
    const int d = threadIdx.x;
    const float inv = __expf(-((float)(2 * d) / (float)HDc) * logf(10000.0f));
    const float f = (float)s * inv;
    const float c = cosf(f), sn = sinf(f);
    const size_t ib = ((size_t)b * Sc + s) * Hc + h * HDc;
    const size_t ob = ((size_t)(b * NHc + h) * Sc + s) * HDc;
    const float qscale = 0.088388347648318447f;
    const float q0 = qf[ib + d], q1 = qf[ib + d + 64];
    const float k0 = kf[ib + d], k1 = kf[ib + d + 64];
    qh[ob + d]      = (_Float16)((q0 * c - q1 * sn) * qscale);
    qh[ob + d + 64] = (_Float16)((q1 * c + q0 * sn) * qscale);
    kh[ob + d]      = (_Float16)(k0 * c - k1 * sn);
    kh[ob + d + 64] = (_Float16)(k1 * c + k0 * sn);
}

__global__ void vtrans_f16_kernel(const float* __restrict__ vf,
                                  _Float16* __restrict__ vT) {
    const int idx = blockIdx.x;
    const int s = idx % Sc;
    const int h = (idx / Sc) % NHc;
    const int b = idx / (Sc * NHc);
    const int d = threadIdx.x;
    vT[((size_t)(b * NHc + h) * HDc + d) * Sc + s] =
        (_Float16)vf[((size_t)b * Sc + s) * Hc + h * HDc + d];
}

union AFrag { v16h v; unsigned int u[8]; };

__global__ __launch_bounds__(256) void flash_attn_wmma_kernel(
    const _Float16* __restrict__ qh, const _Float16* __restrict__ kh,
    const _Float16* __restrict__ vT, float* __restrict__ out) {
    const int bh   = blockIdx.y;
    const int wave = threadIdx.x >> 5;
    const int lane = threadIdx.x & 31;
    const int half_ = lane >> 4;
    const int nl   = lane & 15;
    const int q0   = blockIdx.x * 128 + wave * 16;

    const _Float16* Qb = qh + (size_t)bh * Sc * HDc;
    const _Float16* Kb = kh + (size_t)bh * Sc * HDc;
    const _Float16* Vb = vT + (size_t)bh * HDc * Sc;

    __shared__ __align__(16) _Float16 sP[8][16][32];

    AFrag aQ[4];
#pragma unroll
    for (int cch = 0; cch < 4; ++cch)
#pragma unroll
        for (int j = 0; j < 8; ++j) {
            const int k = ((j & 3) << 1) | ((j >> 2) << 4) | (half_ << 3);
            aQ[cch].u[j] =
                *(const unsigned int*)(Qb + (size_t)(q0 + nl) * HDc + 32 * cch + k);
        }

    v8f o[8];
    float mrow[8], lrow[8];
    const v8f z8 = {0.f, 0.f, 0.f, 0.f, 0.f, 0.f, 0.f, 0.f};
#pragma unroll
    for (int i = 0; i < 8; ++i) { o[i] = z8; mrow[i] = -1e30f; lrow[i] = 0.f; }

    const int qmax = q0 + 15;
    for (int sk0 = 0; sk0 <= qmax; sk0 += 32) {
        v8f s0 = z8, s1 = z8;
#pragma unroll
        for (int cch = 0; cch < 4; ++cch) {
            AFrag bK0, bK1;
#pragma unroll
            for (int j = 0; j < 8; ++j) {
                const int k = ((j & 3) << 1) | ((j >> 2) << 4) | (half_ << 3);
                bK0.u[j] = *(const unsigned int*)(Kb + (size_t)(sk0 + nl) * HDc + 32 * cch + k);
                bK1.u[j] = *(const unsigned int*)(Kb + (size_t)(sk0 + 16 + nl) * HDc + 32 * cch + k);
            }
            s0 = __builtin_amdgcn_wmma_f32_16x16x32_f16(false, aQ[cch].v, false, bK0.v,
                                                        (short)0, s0, false, false);
            s1 = __builtin_amdgcn_wmma_f32_16x16x32_f16(false, aQ[cch].v, false, bK1.v,
                                                        (short)0, s1, false, false);
        }
#pragma unroll
        for (int j = 0; j < 8; ++j) {
            const int rq = q0 + j + (half_ << 3);
            const float v0 = (sk0 + nl <= rq) ? s0[j] : -1e30f;
            const float v1 = (sk0 + 16 + nl <= rq) ? s1[j] : -1e30f;
            float mx = fmaxf(v0, v1);
            for (int off = 1; off < 16; off <<= 1) mx = fmaxf(mx, __shfl_xor(mx, off, 32));
            const float m2 = fmaxf(mrow[j], mx);
            const float p0 = __expf(v0 - m2);
            const float p1 = __expf(v1 - m2);
            float ps = p0 + p1;
            for (int off = 1; off < 16; off <<= 1) ps += __shfl_xor(ps, off, 32);
            const float alpha = __expf(mrow[j] - m2);
            lrow[j] = lrow[j] * alpha + ps;
            mrow[j] = m2;
#pragma unroll
            for (int cch = 0; cch < 8; ++cch) o[cch][j] = o[cch][j] * alpha;
            sP[wave][j + (half_ << 3)][nl]      = (_Float16)p0;
            sP[wave][j + (half_ << 3)][16 + nl] = (_Float16)p1;
        }
        AFrag aP;
#pragma unroll
        for (int j = 0; j < 8; ++j) {
            const int k = ((j & 3) << 1) | ((j >> 2) << 4) | (half_ << 3);
            aP.u[j] = *(const unsigned int*)(&sP[wave][nl][k]);
        }
#pragma unroll
        for (int cch = 0; cch < 8; ++cch) {
            AFrag bV;
#pragma unroll
            for (int j = 0; j < 8; ++j) {
                const int k = ((j & 3) << 1) | ((j >> 2) << 4) | (half_ << 3);
                bV.u[j] = *(const unsigned int*)(Vb + (size_t)(16 * cch + nl) * Sc + sk0 + k);
            }
            o[cch] = __builtin_amdgcn_wmma_f32_16x16x32_f16(false, aP.v, false, bV.v,
                                                            (short)0, o[cch], false, false);
        }
    }

    const int b = bh / NHc, h = bh % NHc;
#pragma unroll
    for (int j = 0; j < 8; ++j) {
        const float invl = 1.0f / lrow[j];
        const int rq = q0 + j + (half_ << 3);
        const size_t base = ((size_t)b * Sc + rq) * Hc + h * HDc;
#pragma unroll
        for (int cch = 0; cch < 8; ++cch)
            out[base + 16 * cch + nl] = o[cch][j] * invl;
    }
}

__global__ void quant_i8_kernel(const float* __restrict__ x,
                                const float* __restrict__ sptr,
                                signed char* __restrict__ out, size_t n) {
    const size_t i = (size_t)blockIdx.x * 256 + threadIdx.x;
    if (i < n) out[i] = quant8(x[i] / (*sptr));
}

__global__ void silu_mul_quant_kernel(const float* __restrict__ g,
                                      const float* __restrict__ u,
                                      const float* __restrict__ sptr,
                                      signed char* __restrict__ out, size_t n) {
    const size_t i = (size_t)blockIdx.x * 256 + threadIdx.x;
    if (i < n) {
        const float gv = g[i];
        const float a = (gv / (1.0f + __expf(-gv))) * u[i];
        out[i] = quant8(a / (*sptr));
    }
}

extern "C" void kernel_launch(void* const* d_in, const int* in_sizes, int n_in,
                              void* d_out, int out_size, void* d_ws, size_t ws_size,
                              hipStream_t stream) {
    const float* hidden = (const float*)d_in[0];
    const float* ln1 = (const float*)d_in[1];
    const float* ln2 = (const float*)d_in[2];
    const signed char* Wq = (const signed char*)d_in[3];
    const signed char* Wk = (const signed char*)d_in[4];
    const signed char* Wv = (const signed char*)d_in[5];
    const signed char* Wo = (const signed char*)d_in[6];
    const signed char* Wg = (const signed char*)d_in[7];
    const signed char* Wu = (const signed char*)d_in[8];
    const signed char* Wd = (const signed char*)d_in[9];
    const float* bq = (const float*)d_in[10];
    const float* bk = (const float*)d_in[11];
    const float* bv = (const float*)d_in[12];
    const float* bo = (const float*)d_in[13];
    const float* bg = (const float*)d_in[14];
    const float* bu = (const float*)d_in[15];
    const float* bd = (const float*)d_in[16];
    const float* sq = (const float*)d_in[17];
    const float* sk = (const float*)d_in[18];
    const float* sv = (const float*)d_in[19];
    const float* so = (const float*)d_in[20];
    const float* sg = (const float*)d_in[21];
    const float* su = (const float*)d_in[22];
    const float* sd = (const float*)d_in[23];
    const float* o_in_scale = (const float*)d_in[24];
    const float* down_in_scale = (const float*)d_in[25];

    char* w = (char*)d_ws;
    size_t off = 0;
    auto take = [&](size_t bytes) -> char* {
        char* p = w + off;
        off += (bytes + 255) & ~(size_t)255;
        return p;
    };
    signed char* A8 = (signed char*)take((size_t)MT * Ic);
    float* F1 = (float*)take((size_t)MT * Ic * 4);
    float* F2 = (float*)take((size_t)MT * Ic * 4);
    float* F3 = (float*)take((size_t)MT * Hc * 4);
    float* F4 = (float*)take((size_t)MT * Hc * 4);
    _Float16* qh = (_Float16*)take((size_t)MT * Hc * 2);
    _Float16* kh = (_Float16*)take((size_t)MT * Hc * 2);
    _Float16* vT = (_Float16*)take((size_t)MT * Hc * 2);

    const dim3 blk(256);
    const dim3 gH(Hc / 64, MT / 64);
    const dim3 gI(Ic / 64, MT / 64);

    rmsnorm_quant_kernel<<<MT, blk, 0, stream>>>(hidden, ln1, A8, Hc);
    gemm_i8_wmma_kernel<<<gH, blk, 0, stream>>>(A8, Wq, sq, bq, nullptr, F1, MT, Hc, Hc);
    gemm_i8_wmma_kernel<<<gH, blk, 0, stream>>>(A8, Wk, sk, bk, nullptr, F2, MT, Hc, Hc);
    gemm_i8_wmma_kernel<<<gH, blk, 0, stream>>>(A8, Wv, sv, bv, nullptr, F3, MT, Hc, Hc);
    rope_f16_kernel<<<Bc * NHc * Sc, 64, 0, stream>>>(F1, F2, qh, kh);
    vtrans_f16_kernel<<<Bc * NHc * Sc, HDc, 0, stream>>>(F3, vT);
    flash_attn_wmma_kernel<<<dim3(Sc / 128, Bc * NHc), blk, 0, stream>>>(qh, kh, vT, F3);
    {
        const size_t n = (size_t)MT * Hc;
        quant_i8_kernel<<<(n + 255) / 256, blk, 0, stream>>>(F3, o_in_scale, A8, n);
    }
    gemm_i8_wmma_kernel<<<gH, blk, 0, stream>>>(A8, Wo, so, bo, hidden, F4, MT, Hc, Hc);
    rmsnorm_quant_kernel<<<MT, blk, 0, stream>>>(F4, ln2, A8, Hc);
    gemm_i8_wmma_kernel<<<gI, blk, 0, stream>>>(A8, Wg, sg, bg, nullptr, F1, MT, Ic, Hc);
    gemm_i8_wmma_kernel<<<gI, blk, 0, stream>>>(A8, Wu, su, bu, nullptr, F2, MT, Ic, Hc);
    {
        const size_t n = (size_t)MT * Ic;
        silu_mul_quant_kernel<<<(n + 255) / 256, blk, 0, stream>>>(F1, F2, down_in_scale, A8, n);
    }
    gemm_i8_wmma_kernel<<<gH, blk, 0, stream>>>(A8, Wd, sd, bd, F4, (float*)d_out, MT, Hc, Ic);
}
